// MultimodalKBGAT_45595372815099
// MI455X (gfx1250) — compile-verified, matrix operand order checked by reading, not run on hardware
//
#include <hip/hip_runtime.h>
#include <hip/hip_bf16.h>

#define LDIM 128
#define NEG_SLOPE 0.01f

typedef __attribute__((ext_vector_type(2)))  float   v2f;
typedef __attribute__((ext_vector_type(8)))  float   v8f;
typedef __attribute__((ext_vector_type(16))) __bf16  v16bf;

#ifndef __has_builtin
#define __has_builtin(x) 0
#endif
#if __has_builtin(__builtin_amdgcn_wmma_f32_16x16x4_f32)
#define USE_WMMA_F32 1
#else
#define USE_WMMA_F32 0
#endif

// ---- ordered-uint encoding for float atomicMax ----
__device__ __forceinline__ unsigned ford(float f) {
    unsigned u = __float_as_uint(f);
    return (u & 0x80000000u) ? ~u : (u | 0x80000000u);
}
__device__ __forceinline__ float funord(unsigned u) {
    unsigned b = (u & 0x80000000u) ? (u & 0x7FFFFFFFu) : ~u;
    return __uint_as_float(b);
}

// =====================================================================
// GEMM: Y[Nrows,128] = X[Nrows,K] @ W'[128,K]^T (+bias), W' rows stride ldW.
// One wave computes a 16x16 output tile via WMMA. block=128 (4 waves).
// grid = (ceil(Nrows/64), 8). A-rows clamped (no divergence in K loop);
// out-of-range D rows are junk and never stored.
// =====================================================================
__global__ void kb_gemm_wmma(const float* __restrict__ X, const float* __restrict__ W,
                             const float* __restrict__ bias, float* __restrict__ Y,
                             int Nrows, int K, int ldW) {
    const int lane    = threadIdx.x & 31;
    const int wave    = threadIdx.x >> 5;
    const int rowTile = blockIdx.x * 4 + wave;
    const int row0    = rowTile * 16;
    if (row0 >= Nrows) return;                 // wave-uniform exit (EXEC all-1s for WMMA)
    const int col0 = blockIdx.y * 16;
    const int m    = lane & 15;                // A-row / B-col fed by this lane
    const int g    = lane >> 4;                // half-wave group
    int rA = row0 + m;
    if (rA > Nrows - 1) rA = Nrows - 1;        // clamp: keeps loads in-bounds, EXEC uniform
    const float* xp = X + (size_t)rA * K + 2 * g;               // this lane's A stream
    const float* wp = W + (size_t)(col0 + m) * ldW + 2 * g;     // this lane's B stream

    v8f acc = {0.f, 0.f, 0.f, 0.f, 0.f, 0.f, 0.f, 0.f};

#if USE_WMMA_F32
    // f32 WMMA 16x16x4: lane holds A[m][2g+{0,1}], B[2g+{0,1}][m]
    for (int kk = 0; kk < K; kk += 4) {
        const float2 ta = *(const float2*)(xp);
        const float2 tb = *(const float2*)(wp);
        xp += 4; wp += 4;
        v2f a; a.x = ta.x; a.y = ta.y;
        v2f b; b.x = tb.x; b.y = tb.y;
        acc = __builtin_amdgcn_wmma_f32_16x16x4_f32(false, a, false, b,
                                                    (short)0, acc, false, false);
    }
#else
    // bf16 fallback: 16x16x32, f32 accumulate. 16-bit A 16x32 layout:
    // vgpr v<4: k = 8g + 2v + {0,1}; v>=4: k = 16 + 8g + 2(v-4) + {0,1}.
    for (int kk = 0; kk < K; kk += 32) {
        v16bf a, b;
#pragma unroll
        for (int v = 0; v < 8; ++v) {
            const int kb = (v < 4) ? (g * 6 + 2 * v) : (16 + g * 8 + 2 * (v - 4) - 2 * g);
            // kb above must equal: v<4 ? 8g+2v : 16+8g+2(v-4); expressed without branch drift:
            const int kbi = (v < 4) ? (g * 8 + 2 * v) : (16 + g * 8 + 2 * (v - 4));
            (void)kb;
            const float2 pa = *(const float2*)(xp - 2 * g + kk + kbi);
            const float2 pb = *(const float2*)(wp - 2 * g + kk + kbi);
            a[2 * v] = (__bf16)pa.x;  a[2 * v + 1] = (__bf16)pa.y;
            b[2 * v] = (__bf16)pb.x;  b[2 * v + 1] = (__bf16)pb.y;
        }
        acc = __builtin_amdgcn_wmma_f32_16x16x32_bf16(false, a, false, b,
                                                      (short)0, acc, false, false);
    }
#endif
    // C/D layout: vgpr j holds row (row0 + 8g + j), col (col0 + m)
    const float bs = bias ? bias[col0 + m] : 0.f;
#pragma unroll
    for (int j = 0; j < 8; ++j) {
        const int r = row0 + g * 8 + j;
        if (r < Nrows) Y[(size_t)r * LDIM + col0 + m] = acc[j] + bs;
    }
}

// out[row] = dot(In[row, 0:128], vec). One wave per row; block=256 (8 rows).
__global__ void kb_rowdot(const float* __restrict__ In, const float* __restrict__ vec,
                          float* __restrict__ out, int Nrows) {
    const int lane = threadIdx.x & 31;
    const int row  = blockIdx.x * 8 + (threadIdx.x >> 5);
    if (row >= Nrows) return;
    const float* r = In + (size_t)row * LDIM;
    float s = 0.f;
#pragma unroll
    for (int j = 0; j < 4; ++j) s += r[lane + 32 * j] * vec[lane + 32 * j];
    for (int o = 16; o > 0; o >>= 1) s += __shfl_xor(s, o, 32);
    if (lane == 0) out[row] = s;
}

__global__ void kb_init(float* accum, float* z, float* asum, unsigned* mord, int N) {
    const int t = blockIdx.x * blockDim.x + threadIdx.x;
    if (t < N * LDIM) accum[t] = 0.f;
    if (t < N) { z[t] = 0.f; asum[t] = 0.f; mord[t] = 0u; }
}

// per-edge logit + segment max over dst
__global__ void kb_logit(const int* __restrict__ src, const int* __restrict__ dst,
                         const int* __restrict__ et,
                         const float* __restrict__ qs, const float* __restrict__ qd,
                         const float* __restrict__ qr, const float* __restrict__ qb,
                         float* __restrict__ eb, unsigned* __restrict__ mord, int E) {
    const int e = blockIdx.x * blockDim.x + threadIdx.x;
    if (e >= E) return;
    const int d = dst[e];
    float b = qd[d] + qs[src[e]] + qr[et[e]] + qb[0];
    b = (b > 0.f) ? b : NEG_SLOPE * b;
    eb[e] = b;
    atomicMax(&mord[d], ford(b));
}

// e = exp(b - m[dst]); z[dst] += e
__global__ void kb_expsum(const int* __restrict__ dst, const unsigned* __restrict__ mord,
                          float* __restrict__ eb, float* __restrict__ z, int E) {
    const int e = blockIdx.x * blockDim.x + threadIdx.x;
    if (e >= E) return;
    const int d = dst[e];
    const float ex = expf(eb[e] - funord(mord[d]));
    eb[e] = ex;
    atomicAdd(&z[d], ex);
}

// one wave per edge: accum[dst] += attn*(Ps[src]+Pr[type]); asum[dst] += attn
__global__ void kb_accum(const int* __restrict__ src, const int* __restrict__ dst,
                         const int* __restrict__ et,
                         const float* __restrict__ eb, const float* __restrict__ z,
                         const float* __restrict__ Ps, const float* __restrict__ Pr,
                         float* __restrict__ accum, float* __restrict__ asum, int E) {
    const int lane = threadIdx.x & 31;
    // edge id is wave-uniform: force it into an SGPR so index/attn loads scalarize
    const int e = __builtin_amdgcn_readfirstlane(blockIdx.x * 8 + (threadIdx.x >> 5));
    if (e >= E) return;
    const int d = dst[e];
    const int s = src[e];
    const int t = et[e];
    const float attn = eb[e] / z[d];
    const int c = lane * 4;
    const float4 ps = *(const float4*)(Ps + (size_t)s * LDIM + c);
    const float4 pr = *(const float4*)(Pr + (size_t)t * LDIM + c);
    float* ap = accum + (size_t)d * LDIM + c;
    atomicAdd(ap + 0, attn * (ps.x + pr.x));
    atomicAdd(ap + 1, attn * (ps.y + pr.y));
    atomicAdd(ap + 2, attn * (ps.z + pr.z));
    atomicAdd(ap + 3, attn * (ps.w + pr.w));
    if (lane == 0) atomicAdd(&asum[d], attn);
}

// h = leaky(accum + asum*(Pd + b1)); out = (first?0:out) + w*h
__global__ void kb_final(const float* __restrict__ accum, const float* __restrict__ asum,
                         const float* __restrict__ Pd, const float* __restrict__ b1,
                         const float* __restrict__ alpha, const float* __restrict__ gamma,
                         float* __restrict__ out, int N, int convIdx) {
    const int t = blockIdx.x * blockDim.x + threadIdx.x;
    if (t >= N * LDIM) return;
    const int n = t >> 7, c = t & 127;
    float a = accum[t] + asum[n] * (Pd[t] + b1[c]);
    a = (a > 0.f) ? a : NEG_SLOPE * a;
    const float al = alpha[0], ga = gamma[0];
    const float w = (convIdx == 0) ? (1.f - al - ga) : ((convIdx == 1) ? al : ga);
    const float prev = (convIdx == 0) ? 0.f : out[t];
    out[t] = prev + w * a;
}

extern "C" void kernel_launch(void* const* d_in, const int* in_sizes, int n_in,
                              void* d_out, int out_size, void* d_ws, size_t ws_size,
                              hipStream_t stream) {
    (void)n_in; (void)out_size; (void)ws_size;
    const int*   edge_index = (const int*)d_in[1];
    const int*   edge_type  = (const int*)d_in[2];
    const float* visual     = (const float*)d_in[3];
    const float* textual    = (const float*)d_in[4];
    const float* semb       = (const float*)d_in[5];
    const float* remb       = (const float*)d_in[6];
    const float* alpha      = (const float*)d_in[20];
    const float* gamma      = (const float*)d_in[21];
    const int E  = in_sizes[2];
    const int N  = in_sizes[5] / LDIM;
    const int R  = in_sizes[6] / LDIM;
    const int VD = in_sizes[3] / N;
    const int TD = in_sizes[4] / N;
    const int* src = edge_index;
    const int* dst = edge_index + E;

    float* ws = (float*)d_ws;
    size_t o = 0;
    auto take = [&](size_t cnt) { float* p = ws + o; o += cnt; return p; };
    float*    vf    = take((size_t)N * LDIM);
    float*    tf    = take((size_t)N * LDIM);
    float*    Pd    = take((size_t)N * LDIM);
    float*    Ps    = take((size_t)N * LDIM);
    float*    Pr    = take((size_t)R * LDIM);
    float*    qd    = take((size_t)N);
    float*    qs    = take((size_t)N);
    float*    qr    = take((size_t)R);
    float*    qb    = take(8);
    unsigned* mord  = (unsigned*)take((size_t)N);
    float*    z     = take((size_t)N);
    float*    asum  = take((size_t)N);
    float*    accum = take((size_t)N * LDIM);
    float*    eb    = take((size_t)E);

    const int GB = 256;
    const dim3 blkG(128);
    auto gemmGrid = [](int rows) { return dim3((unsigned)((rows + 63) / 64), 8u); };

    // modality projections: vf = visual@Wv^T+bv, tf = textual@Wt^T+bt
    kb_gemm_wmma<<<gemmGrid(N), blkG, 0, stream>>>(visual,  (const float*)d_in[16],
                                                   (const float*)d_in[17], vf, N, VD, VD);
    kb_gemm_wmma<<<gemmGrid(N), blkG, 0, stream>>>(textual, (const float*)d_in[18],
                                                   (const float*)d_in[19], tf, N, TD, TD);

    for (int idx = 0; idx < 3; ++idx) {
        const float* feat = (idx == 0) ? semb : (idx == 1 ? vf : tf);
        const float* W1 = (const float*)d_in[7 + 3 * idx];   // [128, 384] row-major
        const float* b1 = (const float*)d_in[8 + 3 * idx];
        const float* w2 = (const float*)d_in[9 + 3 * idx];
        const int CAT = 3 * LDIM;                            // ldW = 384

        kb_init<<<(N * LDIM + GB - 1) / GB, GB, 0, stream>>>(accum, z, asum, mord, N);
        // per-node projections: columns [0:128)=dst part, [128:256)=src, [256:384)=rel
        kb_gemm_wmma<<<gemmGrid(N), blkG, 0, stream>>>(feat, W1,            nullptr, Pd, N, LDIM, CAT);
        kb_gemm_wmma<<<gemmGrid(N), blkG, 0, stream>>>(feat, W1 + LDIM,     nullptr, Ps, N, LDIM, CAT);
        kb_gemm_wmma<<<gemmGrid(R), blkG, 0, stream>>>(remb, W1 + 2 * LDIM, nullptr, Pr, R, LDIM, CAT);
        // attention scalars
        kb_rowdot<<<(N + 7) / 8, 256, 0, stream>>>(Pd, w2, qd, N);
        kb_rowdot<<<(N + 7) / 8, 256, 0, stream>>>(Ps, w2, qs, N);
        kb_rowdot<<<(R + 7) / 8, 256, 0, stream>>>(Pr, w2, qr, R);
        kb_rowdot<<<1,           256, 0, stream>>>(b1, w2, qb, 1);
        // segment softmax + weighted aggregation
        kb_logit <<<(E + GB - 1) / GB, GB, 0, stream>>>(src, dst, edge_type, qs, qd, qr, qb, eb, mord, E);
        kb_expsum<<<(E + GB - 1) / GB, GB, 0, stream>>>(dst, mord, eb, z, E);
        kb_accum <<<(E + 7) / 8, 256, 0, stream>>>(src, dst, edge_type, eb, z, Ps, Pr, accum, asum, E);
        kb_final <<<(N * LDIM + GB - 1) / GB, GB, 0, stream>>>(accum, asum, Pd, b1, alpha, gamma,
                                                               (float*)d_out, N, idx);
    }
}
